// GLCMLayer_76922864271867
// MI455X (gfx1250) — 8-bit matrix operand order repaired and checked by reading, compiled, not run on hardware
//
#include <hip/hip_runtime.h>
#include <hip/hip_bf16.h>
#include <math.h>

typedef int v8i __attribute__((ext_vector_type(8)));

__global__ void glcm_quantize(const float* __restrict__ x,
                              unsigned char* __restrict__ gl, int n) {
    int i = blockIdx.x * blockDim.x + threadIdx.x;
    if (i < n) {
        float v = x[i];
        int g = (int)ceilf(v * 0.03125f) - 1;
        gl[i] = (g >= 0 && g <= 7) ? (unsigned char)g : (unsigned char)0xFF;
    }
}

__device__ __forceinline__ unsigned eq_bytes(unsigned q, unsigned m4) {
    unsigned x = q ^ m4;
    unsigned r = ~(((x & 0x7F7F7F7Fu) + 0x7F7F7F7Fu) | x | 0x7F7F7F7Fu);
    return r >> 7;
}

__global__ __launch_bounds__(256) void glcm_cell(
        const unsigned char* __restrict__ gl, float* __restrict__ out) {
    __shared__ unsigned char tile[38 * 40];
    __shared__ unsigned int  stage[8][2][16];

    const int blk = blockIdx.x;
    const int b = blk >> 6;
    const int Y = (blk >> 3) & 7;
    const int X = blk & 7;
    const int Y0 = Y << 5, X0 = X << 5;
    const int R0 = Y0 - 3, C0 = X0 - 3;
    const unsigned char* glb = gl + b * 65536;

    __builtin_prefetch(glb + min(max(R0, 0), 255) * 256, 0, 0);

    for (int idx = threadIdx.x; idx < 38 * 38; idx += 256) {
        int r = idx / 38, c = idx - r * 38;
        int gy = min(max(R0 + r, 0), 255);
        int gx = min(max(C0 + c, 0), 255);
        tile[r * 40 + c] = glb[gy * 256 + gx];
    }
    __syncthreads();

    const int wave = threadIdx.x >> 5;
    const int lane = threadIdx.x & 31;
    const int m = lane & 15;
    const unsigned m4 = (unsigned)m * 0x01010101u;

    const int DXT[8] = {1, 1, 0, -1, -1, -1, 0, 1};
    const int DYT[8] = {0, -1, -1, -1, 0, 1, 1, 1};
    const int dx = DXT[wave], dy = DYT[wave];

    const int baseA = (lane < 16) ? 0 : 2;
    const int baseB = (lane < 16) ? 0 : 2;
    const int idxA[8] = {0, 1, 4, 5, 8, 9, 12, 13};
    const int idxB[8] = {0, 1, 4, 5, 8, 9, 12, 13};

    v8i acc = {0, 0, 0, 0, 0, 0, 0, 0};

    unsigned* s1 = &stage[wave][0][0];
    unsigned* s2 = &stage[wave][1][0];

    for (int k = 0; k < 400; ++k) {
        const int t = k >> 4;
        const int u = t / 5 - 2, v = t % 5 - 2;
        unsigned p1 = 0, p2 = 0;
#pragma unroll
        for (int i = 0; i < 2; ++i) {
            int p  = ((k & 15) << 6) | (lane << 1) | i;
            int gy = Y0 + (p >> 5) + u;
            int gx = X0 + (p & 31) + v;
            int ry = gy < 0 ? -gy : (gy > 255 ? 510 - gy : gy);
            int rx = gx < 0 ? -gx : (gx > 255 ? 510 - gx : gx);
            unsigned c1 = tile[(ry - R0) * 40 + (rx - C0)];
            int y2 = min(max(ry + dy, 0), 255);
            int x2 = min(max(rx + dx, 0), 255);
            unsigned c2 = tile[(y2 - R0) * 40 + (x2 - C0)];
            p1 |= c1 << (8 * i);
            p2 |= c2 << (8 * i);
        }
        ((unsigned short*)s1)[lane] = (unsigned short)p1;
        ((unsigned short*)s2)[lane] = (unsigned short)p2;
        asm volatile("" ::: "memory");

        v8i A, B;
#pragma unroll
        for (int j = 0; j < 8; ++j) {
            A[j] = (int)eq_bytes(s1[baseA + idxA[j]], m4);
            B[j] = (int)eq_bytes(s2[baseB + idxB[j]], m4);
        }
        acc = __builtin_amdgcn_wmma_i32_16x16x64_iu8(false, A, false, B, acc,
                                                     false, false);
        asm volatile("" ::: "memory");
    }

    if (lane < 8) {
        const float s = 1.0f / 1024.0f;
#pragma unroll
        for (int j = 0; j < 8; ++j) {
            int ch = wave * 64 + j * 8 + lane;
            out[((b * 512 + ch) * 8 + Y) * 8 + X] = (float)acc[j] * s;
        }
    }
}

extern "C" void kernel_launch(void* const* d_in, const int* in_sizes, int n_in,
                              void* d_out, int out_size, void* d_ws, size_t ws_size,
                              hipStream_t stream) {
    const float* x = (const float*)d_in[0];
    float* out = (float*)d_out;
    unsigned char* glmap = (unsigned char*)d_ws;
    int n = in_sizes[0];
    glcm_quantize<<<(n + 255) / 256, 256, 0, stream>>>(x, glmap, n);
    glcm_cell<<<128, 256, 0, stream>>>(glmap, out);
}
